// GATMamba_57226144252521
// MI455X (gfx1250) — matrix operand order repaired and checked by reading, not compiled, not run on hardware
//
#include <hip/hip_runtime.h>
#include <hip/hip_bf16.h>

// ---------------------------------------------------------------------------
// Model constants (from reference)
// ---------------------------------------------------------------------------
#define NN      16384      // nodes
#define BB      8          // batches
#define LL      2048       // seq len per batch
#define EE      262144     // edges
#define CC      512        // channels (UNI 496 + PE 16)
#define UNI     496
#define SS      16         // SSM state
#define KCONV   4
#define DTR     32

typedef __bf16 bf16t;
typedef __attribute__((ext_vector_type(16))) __bf16 v16bf;
typedef __attribute__((ext_vector_type(8)))  float  v8f;

enum { ACT_NONE = 0, ACT_RELU = 1, ACT_SILU = 2, ACT_SOFTPLUS = 3 };

// ---------------------------------------------------------------------------
// Generic WMMA GEMM:  C[M,N] = act( A[M,K] @ B[K,N] + bias + addsrc )
// Requirements (true for every call site here): M % 128 == 0, K % 32 == 0.
// N may be ragged (N=496): handled by a block-uniform guard on B staging and
// per-column guard in the epilogue.
// bf16 inputs (converted on LDS stage-in), f32 accumulation via
// v_wmma_f32_16x16x32_bf16.  Block = 256 threads = 8 waves, tile 128x64.
// Software-pipelined: tile k0+32 is prefetched into VGPRs while WMMAs for
// tile k0 execute out of LDS.
// ---------------------------------------------------------------------------
#define TM 128
#define TN 64
#define KB 32
#define LPAD 4   // bf16 pad -> 72B row stride in LDS (conflict-free frag reads)

__global__ __launch_bounds__(256)
void wmma_gemm(const float* __restrict__ A, int lda,
               const float* __restrict__ Bw, int ldb,
               float* __restrict__ C, int ldc,
               const float* __restrict__ bias,
               const float* __restrict__ addsrc, int ldadd,
               int M, int N, int K, int act)
{
    __shared__ bf16t As[TM][KB + LPAD];
    __shared__ bf16t Bs[TN][KB + LPAD];

    const int n0   = blockIdx.x * TN;
    const int m0   = blockIdx.y * TM;
    const int tid  = threadIdx.x;
    const int wave = tid >> 5;
    const int lane = tid & 31;
    const int lmod = lane & 15;
    const int kh   = lane >> 4;

    // block-uniform: does this N-tile need column guarding?
    const bool nguard = (n0 + TN > N);

    // staging register file (software pipeline)
    float aReg[16];   // 128x32 / 256 thr
    float bReg[8];    // 32x64  / 256 thr
    // A map: idx = p*256+tid -> row idx>>5, col idx&31   (tid-contig = col-contig)
    const int ar = tid >> 5;          // +8 per p
    const int ac = tid & 31;
    // B map: idx = p*256+tid -> krow idx>>6, ncol idx&63 (tid-contig = col-contig)
    const int bk = tid >> 6;          // +4 per p
    const int bn = tid & 63;

    auto loadA = [&](int k0) {
        const float* ap = A + (size_t)(m0 + ar) * lda + (k0 + ac);
        #pragma unroll
        for (int p = 0; p < 16; ++p) aReg[p] = ap[(size_t)(p * 8) * lda];
    };
    auto loadB = [&](int k0) {
        const float* bp = Bw + (size_t)(k0 + bk) * ldb + (n0 + bn);
        if (!nguard) {
            #pragma unroll
            for (int p = 0; p < 8; ++p) bReg[p] = bp[(size_t)(p * 4) * ldb];
        } else {
            #pragma unroll
            for (int p = 0; p < 8; ++p)
                bReg[p] = (n0 + bn < N) ? bp[(size_t)(p * 4) * ldb] : 0.f;
        }
    };

    v8f acc[4] = {};
    loadA(0);
    loadB(0);

    for (int k0 = 0; k0 < K; k0 += KB) {
        // ---- spill staged registers into LDS (f32 -> bf16) ----
        #pragma unroll
        for (int p = 0; p < 16; ++p) As[ar + p * 8][ac] = (bf16t)aReg[p];
        #pragma unroll
        for (int p = 0; p < 8; ++p)  Bs[bn][bk + p * 4] = (bf16t)bReg[p];
        __syncthreads();

        // ---- prefetch next tile into registers (overlaps WMMA phase) ----
        if (k0 + KB < K) { loadA(k0 + KB); loadB(k0 + KB); }

        // ---- A fragment: 16x32 bf16, ISA 7.12.2 layout ----
        union { unsigned u[8]; v16bf v; } af;
        const int mrow = wave * 16 + lmod;
        #pragma unroll
        for (int vv = 0; vv < 8; ++vv) {
            int kb = (vv < 4) ? (kh * 8 + vv * 2) : (16 + kh * 8 + (vv - 4) * 2);
            af.u[vv] = *(const unsigned*)&As[mrow][kb];
        }
        // ---- all 4 B fragments first, then the WMMA chain ----
        union { unsigned u[8]; v16bf v; } bfv[4];
        #pragma unroll
        for (int j = 0; j < 4; ++j) {
            const int nrow = j * 16 + lmod;
            #pragma unroll
            for (int vv = 0; vv < 8; ++vv) {
                int kb = (vv < 4) ? (kh * 8 + vv * 2) : (16 + kh * 8 + (vv - 4) * 2);
                bfv[j].u[vv] = *(const unsigned*)&Bs[nrow][kb];
            }
        }
        #pragma unroll
        for (int j = 0; j < 4; ++j)
            acc[j] = __builtin_amdgcn_wmma_f32_16x16x32_bf16(
                false, af.v, false, bfv[j].v, (short)0, acc[j], false, false);
        __syncthreads();
    }

    // ---- epilogue: bias / residual / activation, C/D layout per ISA ----
    #pragma unroll
    for (int j = 0; j < 4; ++j) {
        int n = n0 + j * 16 + lmod;
        if (n >= N) continue;
        float bv = bias ? bias[n] : 0.f;
        #pragma unroll
        for (int vv = 0; vv < 8; ++vv) {
            int m = m0 + wave * 16 + vv + 8 * kh;
            float r = acc[j][vv] + bv;
            if (addsrc) r += addsrc[(size_t)m * ldadd + n];
            if (act == ACT_RELU)          r = fmaxf(r, 0.f);
            else if (act == ACT_SILU)     r = r / (1.f + __expf(-r));
            else if (act == ACT_SOFTPLUS) r = (r > 20.f) ? r : log1pf(__expf(r));
            C[(size_t)m * ldc + n] = r;
        }
    }
}

// ---------------------------------------------------------------------------
// Sinusoidal positional encoding into h0[:, 496:512]
// ---------------------------------------------------------------------------
__global__ __launch_bounds__(256)
void k_pe(const float* __restrict__ x, float* __restrict__ h0)
{
    int i = blockIdx.x * 256 + threadIdx.x;
    if (i >= NN) return;
    float p0 = x[(size_t)i * 1026 + 0];
    float p1 = x[(size_t)i * 1026 + 1];
    float* o = h0 + (size_t)i * CC + UNI;
    #pragma unroll
    for (int j = 0; j < 4; ++j) {
        float denom = __powf(10000.f, 2.f * (float)j / 8.f);
        float a0 = p0 / denom, a1 = p1 / denom;
        o[2 * j + 0] = __sinf(a0);  o[2 * j + 1] = __cosf(a0);
        o[8 + 2 * j] = __sinf(a1);  o[9 + 2 * j] = __cosf(a1);
    }
}

// ---------------------------------------------------------------------------
// a_s / a_d : row-dot of xh with attention vectors (one wave per row)
// ---------------------------------------------------------------------------
__global__ __launch_bounds__(256)
void k_att(const float* __restrict__ xh, const float* __restrict__ atts,
           const float* __restrict__ attd, float* __restrict__ a_s,
           float* __restrict__ a_d)
{
    int row  = blockIdx.x * 8 + (threadIdx.x >> 5);
    int lane = threadIdx.x & 31;
    if (row >= NN) return;
    const float* p = xh + (size_t)row * CC;
    float s = 0.f, d = 0.f;
    for (int c = lane; c < CC; c += 32) {
        float v = p[c]; s += v * atts[c]; d += v * attd[c];
    }
    #pragma unroll
    for (int o = 16; o; o >>= 1) {
        s += __shfl_down(s, o, 32);
        d += __shfl_down(d, o, 32);
    }
    if (lane == 0) { a_s[row] = s; a_d[row] = d; }
}

// order-preserving float<->uint map for atomic segment-max
__device__ __forceinline__ unsigned fenc(float f) {
    unsigned u = __float_as_uint(f);
    return (u & 0x80000000u) ? ~u : (u | 0x80000000u);
}
__device__ __forceinline__ float fdec(unsigned k) {
    unsigned u = (k & 0x80000000u) ? (k & 0x7FFFFFFFu) : ~k;
    return __uint_as_float(u);
}

__global__ __launch_bounds__(256)
void k_gat_init(unsigned* __restrict__ amax, float* __restrict__ den,
                float* __restrict__ gat, const float* __restrict__ b_gat)
{
    size_t i = (size_t)blockIdx.x * 256 + threadIdx.x;
    if (i < (size_t)NN * CC) gat[i] = b_gat[i % CC];
    if (i < NN) { amax[i] = 0x007FFFFFu; /* enc(-inf) */ den[i] = 0.f; }
}

__global__ __launch_bounds__(256)
void k_gat_alpha(const int* __restrict__ src, const int* __restrict__ dst,
                 const float* __restrict__ a_s, const float* __restrict__ a_d,
                 float* __restrict__ ew, unsigned* __restrict__ amax)
{
    int e = blockIdx.x * 256 + threadIdx.x;
    if (e >= EE + NN) return;
    int s = (e < EE) ? src[e] : (e - EE);
    int d = (e < EE) ? dst[e] : (e - EE);
    float al = a_s[s] + a_d[d];
    al = (al > 0.f) ? al : 0.2f * al;           // leaky_relu 0.2
    ew[e] = al;
    atomicMax(&amax[d], fenc(al));
}

__global__ __launch_bounds__(256)
void k_gat_ea(const int* __restrict__ dst, float* __restrict__ ew,
              const unsigned* __restrict__ amax, float* __restrict__ den)
{
    int e = blockIdx.x * 256 + threadIdx.x;
    if (e >= EE + NN) return;
    int d = (e < EE) ? dst[e] : (e - EE);
    float ea = __expf(ew[e] - fdec(amax[d]));
    ew[e] = ea;
    atomicAdd(&den[d], ea);
}

// one wave per edge: gat[dst] += w * xh[src]  (512 feats)
__global__ __launch_bounds__(256)
void k_gat_scatter(const int* __restrict__ src, const int* __restrict__ dst,
                   const float* __restrict__ ew, const float* __restrict__ den,
                   const float* __restrict__ xh, float* __restrict__ gat)
{
    int e    = blockIdx.x * 8 + (threadIdx.x >> 5);
    int lane = threadIdx.x & 31;
    if (e >= EE + NN) return;
    int s = (e < EE) ? src[e] : (e - EE);
    int d = (e < EE) ? dst[e] : (e - EE);
    float w = ew[e] / (den[d] + 1e-16f);
    const float* xs = xh + (size_t)s * CC;
    float* gd = gat + (size_t)d * CC;
    for (int c = lane; c < CC; c += 32) atomicAdd(&gd[c], w * xs[c]);
}

// ---------------------------------------------------------------------------
// BatchNorm stats (block per feature) and fused apply
// ---------------------------------------------------------------------------
__global__ __launch_bounds__(256)
void k_bnstats(const float* __restrict__ h, float* __restrict__ mu,
               float* __restrict__ var)
{
    int f = blockIdx.x;
    float s = 0.f, s2 = 0.f;
    for (int r = threadIdx.x; r < NN; r += 256) {
        float v = h[(size_t)r * CC + f];
        s += v; s2 += v * v;
    }
    __shared__ float sh[256], sh2[256];
    sh[threadIdx.x] = s; sh2[threadIdx.x] = s2;
    __syncthreads();
    for (int o = 128; o; o >>= 1) {
        if (threadIdx.x < o) {
            sh[threadIdx.x]  += sh[threadIdx.x + o];
            sh2[threadIdx.x] += sh2[threadIdx.x + o];
        }
        __syncthreads();
    }
    if (threadIdx.x == 0) {
        float m = sh[0] * (1.f / NN);
        mu[f]  = m;
        var[f] = sh2[0] * (1.f / NN) - m * m;
    }
}

__global__ __launch_bounds__(256)
void k_bnapply(const float* __restrict__ hin, const float* __restrict__ mu,
               const float* __restrict__ var, const float* __restrict__ g,
               const float* __restrict__ b, const float* __restrict__ addsrc,
               int do_relu, float* __restrict__ outp)
{
    size_t i = (size_t)blockIdx.x * 256 + threadIdx.x;
    if (i >= (size_t)NN * CC) return;
    int f = (int)(i % CC);
    float v = (hin[i] - mu[f]) * rsqrtf(var[f] + 1e-5f) * g[f] + b[f];
    if (addsrc) v += addsrc[i];
    if (do_relu) v = fmaxf(v, 0.f);
    outp[i] = v;
}

// ---------------------------------------------------------------------------
// Causal depthwise conv (K=4) + SiLU:  xc = silu(b + sum_k xm[l-3+k]*w[c,k])
// xm = xz[:, :512] (ld 1024)
// ---------------------------------------------------------------------------
__global__ __launch_bounds__(256)
void k_conv(const float* __restrict__ xz, const float* __restrict__ w_conv,
            const float* __restrict__ b_conv, float* __restrict__ xc)
{
    size_t i = (size_t)blockIdx.x * 256 + threadIdx.x;
    if (i >= (size_t)NN * CC) return;
    int c = (int)(i % CC);
    size_t row = i / CC;
    int l = (int)(row % LL);
    float a = b_conv[c];
    #pragma unroll
    for (int k = 0; k < KCONV; ++k) {
        int lk = l - (KCONV - 1) + k;
        if (lk >= 0) a += xz[(row - (KCONV - 1) + k) * (2 * CC) + c] * w_conv[c * KCONV + k];
    }
    xc[i] = a / (1.f + __expf(-a));
}

// ---------------------------------------------------------------------------
// Mamba selective scan, one thread per (batch, channel); fuse (y+D*x)*silu(z)
// ---------------------------------------------------------------------------
__global__ __launch_bounds__(256)
void k_ssm(const float* __restrict__ xc, const float* __restrict__ dt,
           const float* __restrict__ proj, const float* __restrict__ A_log,
           const float* __restrict__ D_p, const float* __restrict__ xz,
           float* __restrict__ yw)
{
    int t = blockIdx.x * 256 + threadIdx.x;
    if (t >= BB * CC) return;
    int b = t >> 9, c = t & (CC - 1);
    float Ac[SS], st[SS];
    #pragma unroll
    for (int s = 0; s < SS; ++s) {
        Ac[s] = -__expf(A_log[c * SS + s]);
        st[s] = 0.f;
    }
    float Dv = D_p[c];
    size_t base = (size_t)b * LL;
    for (int l = 0; l < LL; ++l) {
        size_t row = base + l;
        float dtt = dt[row * CC + c];
        float xt  = xc[row * CC + c];
        const float* pr = proj + row * 64;
        float dx = dtt * xt;
        float y = 0.f;
        #pragma unroll
        for (int s = 0; s < SS; ++s) {
            st[s] = st[s] * __expf(dtt * Ac[s]) + dx * pr[DTR + s];
            y += st[s] * pr[DTR + SS + s];
        }
        float zv = xz[row * (2 * CC) + CC + c];
        float sz = zv / (1.f + __expf(-zv));
        yw[row * CC + c] = (y + Dv * xt) * sz;
    }
}

// ---------------------------------------------------------------------------
// Mean pool over L per (batch, channel)
// ---------------------------------------------------------------------------
__global__ __launch_bounds__(256)
void k_pool(const float* __restrict__ out3, float* __restrict__ pooled)
{
    int t = blockIdx.x * 256 + threadIdx.x;
    if (t >= BB * CC) return;
    int b = t >> 9, c = t & (CC - 1);
    float s = 0.f;
    for (int l = 0; l < LL; ++l)
        s += out3[((size_t)b * LL + l) * CC + c];
    pooled[t] = s * (1.f / (float)LL);
}

// ---------------------------------------------------------------------------
// Tiny MLP head (M=8), single block
// ---------------------------------------------------------------------------
__global__ __launch_bounds__(256)
void k_head(const float* __restrict__ pooled,
            const float* __restrict__ Wf1, const float* __restrict__ bf1,
            const float* __restrict__ Wf2, const float* __restrict__ bf2,
            const float* __restrict__ Wf3, const float* __restrict__ bf3,
            float* __restrict__ pred)
{
    __shared__ float h1s[BB * 256];
    __shared__ float h2s[BB * 128];
    int o = threadIdx.x;
    for (int b = 0; b < BB; ++b) {
        float a = bf1[o];
        for (int k = 0; k < CC; ++k) a += pooled[b * CC + k] * Wf1[k * 256 + o];
        h1s[b * 256 + o] = fmaxf(a, 0.f);
    }
    __syncthreads();
    if (o < 128) {
        for (int b = 0; b < BB; ++b) {
            float a = bf2[o];
            for (int k = 0; k < 256; ++k) a += h1s[b * 256 + k] * Wf2[k * 128 + o];
            h2s[b * 128 + o] = fmaxf(a, 0.f);
        }
    }
    __syncthreads();
    if (o < BB) {
        float a = bf3[0];
        for (int k = 0; k < 128; ++k) a += h2s[o * 128 + k] * Wf3[k];
        pred[o] = a;
    }
}

// ---------------------------------------------------------------------------
// Host-side launch
// ---------------------------------------------------------------------------
extern "C" void kernel_launch(void* const* d_in, const int* in_sizes, int n_in,
                              void* d_out, int out_size, void* d_ws, size_t ws_size,
                              hipStream_t stream)
{
    (void)in_sizes; (void)n_in; (void)out_size; (void)ws_size;

    const float* x       = (const float*)d_in[0];
    const int*   ei      = (const int*)  d_in[1];   // [2,E]
    const float* W_uni   = (const float*)d_in[5];
    const float* b_uni   = (const float*)d_in[6];
    const float* W_gat   = (const float*)d_in[10];
    const float* att_src = (const float*)d_in[11];
    const float* att_dst = (const float*)d_in[12];
    const float* b_gat   = (const float*)d_in[13];
    const float* g1  = (const float*)d_in[14];
    const float* be1 = (const float*)d_in[15];
    const float* g2  = (const float*)d_in[16];
    const float* be2 = (const float*)d_in[17];
    const float* g3  = (const float*)d_in[18];
    const float* be3 = (const float*)d_in[19];
    const float* W_in   = (const float*)d_in[20];
    const float* w_conv = (const float*)d_in[21];
    const float* b_conv = (const float*)d_in[22];
    const float* W_xp   = (const float*)d_in[23];
    const float* W_dt   = (const float*)d_in[24];
    const float* b_dt   = (const float*)d_in[25];
    const float* A_log  = (const float*)d_in[26];
    const float* D_p    = (const float*)d_in[27];
    const float* W_out  = (const float*)d_in[28];
    const float* W_m1   = (const float*)d_in[29];
    const float* b_m1   = (const float*)d_in[30];
    const float* W_m2   = (const float*)d_in[31];
    const float* b_m2   = (const float*)d_in[32];
    const float* Wf1 = (const float*)d_in[33];
    const float* bf1 = (const float*)d_in[34];
    const float* Wf2 = (const float*)d_in[35];
    const float* bf2 = (const float*)d_in[36];
    const float* Wf3 = (const float*)d_in[37];
    const float* bf3 = (const float*)d_in[38];

    const int* e_src = ei;
    const int* e_dst = ei + EE;

    // workspace layout (floats), with aliasing for later stages
    float* ws = (float*)d_ws;
    const size_t SZ = (size_t)NN * CC;          // 8.4M floats
    float* h0   = ws;                            // N x 512
    float* xh   = h0   + SZ;                     // N x 512
    float* gat  = xh   + SZ;                     // N x 512  (becomes h1)
    float* xz   = gat  + SZ;                     // N x 1024
    float* xc   = xz   + 2 * SZ;                 // N x 512
    float* proj = xc   + SZ;                     // N x 64
    float* dtb  = proj + (size_t)NN * 64;        // N x 512
    float* smallb = dtb + SZ;
    float*    a_s  = smallb;                     // N
    float*    a_d  = a_s + NN;                   // N
    float*    ew   = a_d + NN;                   // E+N
    unsigned* amax = (unsigned*)(ew + (EE + NN));// N
    float*    den  = (float*)(amax + NN);        // N
    float*    mu   = den + NN;                   // 512
    float*    var  = mu + CC;                    // 512
    // aliases (lifetimes verified against the dataflow)
    float* yw   = h0;    // written by SSM after h0's last read (xz GEMM)
    float* y2   = xh;    // written after xh's last read (GAT scatter)
    float* outb = xc;    // written after xc's last read (SSM)
    float* m1   = xz;    // written after xz's last read (SSM z-gate)
    float* out2 = dtb;   // written after dt's last read (SSM)

    float* pooled = (float*)d_out + BB;          // output: [pred(8) | pooled(8x512)]
    float* pred   = (float*)d_out;

    const dim3 B256(256);
    const int ROWT = (NN * CC) / 256;            // 32768 elementwise blocks
    const int GEMM_MY = NN / TM;                 // 128 row-tiles

    // 1) positional encoding -> h0[:,496:512]
    k_pe<<<NN / 256, B256, 0, stream>>>(x, h0);
    // 2) xf = x[:,2:1026] @ W_uni + b_uni -> h0[:,:496]
    wmma_gemm<<<dim3((UNI + TN - 1) / TN, GEMM_MY), B256, 0, stream>>>(
        x + 2, 1026, W_uni, UNI, h0, CC, b_uni, nullptr, 0, NN, UNI, 1024, ACT_NONE);
    // 3) xh = h0 @ W_gat
    wmma_gemm<<<dim3(CC / TN, GEMM_MY), B256, 0, stream>>>(
        h0, CC, W_gat, CC, xh, CC, nullptr, nullptr, 0, NN, CC, CC, ACT_NONE);
    // 4) attention logits
    k_att<<<NN / 8, B256, 0, stream>>>(xh, att_src, att_dst, a_s, a_d);
    // 5) GAT softmax-aggregate (with self loops)
    k_gat_init<<<ROWT, B256, 0, stream>>>(amax, den, gat, b_gat);
    k_gat_alpha<<<(EE + NN) / 256, B256, 0, stream>>>(e_src, e_dst, a_s, a_d, ew, amax);
    k_gat_ea<<<(EE + NN) / 256, B256, 0, stream>>>(e_dst, ew, amax, den);
    k_gat_scatter<<<(EE + NN) / 8, B256, 0, stream>>>(e_src, e_dst, ew, den, xh, gat);
    // 6) h1 = BN1(gat)  (in place)
    k_bnstats<<<CC, B256, 0, stream>>>(gat, mu, var);
    k_bnapply<<<ROWT, B256, 0, stream>>>(gat, mu, var, g1, be1, nullptr, 0, gat);
    // 7) xz = h0 @ W_in
    wmma_gemm<<<dim3(2 * CC / TN, GEMM_MY), B256, 0, stream>>>(
        h0, CC, W_in, 2 * CC, xz, 2 * CC, nullptr, nullptr, 0, NN, 2 * CC, CC, ACT_NONE);
    // 8) xc = silu(causal_conv(xm))
    k_conv<<<ROWT, B256, 0, stream>>>(xz, w_conv, b_conv, xc);
    // 9) proj = xc @ W_xp
    wmma_gemm<<<dim3(1, GEMM_MY), B256, 0, stream>>>(
        xc, CC, W_xp, 64, proj, 64, nullptr, nullptr, 0, NN, 64, CC, ACT_NONE);
    // 10) dt = softplus(proj[:, :32] @ W_dt + b_dt)
    wmma_gemm<<<dim3(CC / TN, GEMM_MY), B256, 0, stream>>>(
        proj, 64, W_dt, CC, dtb, CC, b_dt, nullptr, 0, NN, CC, DTR, ACT_SOFTPLUS);
    // 11) selective scan -> yw = (ys + D*xc) * silu(z)
    k_ssm<<<BB * CC / 256, B256, 0, stream>>>(xc, dtb, proj, A_log, D_p, xz, yw);
    // 12) y2 = yw @ W_out
    wmma_gemm<<<dim3(CC / TN, GEMM_MY), B256, 0, stream>>>(
        yw, CC, W_out, CC, y2, CC, nullptr, nullptr, 0, NN, CC, CC, ACT_NONE);
    // 13) out = h1 + BN2(y2)
    k_bnstats<<<CC, B256, 0, stream>>>(y2, mu, var);
    k_bnapply<<<ROWT, B256, 0, stream>>>(y2, mu, var, g2, be2, gat, 0, outb);
    // 14) m1 = relu(out @ W_m1 + b_m1)
    wmma_gemm<<<dim3(2 * CC / TN, GEMM_MY), B256, 0, stream>>>(
        outb, CC, W_m1, 2 * CC, m1, 2 * CC, b_m1, nullptr, 0, NN, 2 * CC, CC, ACT_RELU);
    // 15) out2 = out + m1 @ W_m2 + b_m2
    wmma_gemm<<<dim3(CC / TN, GEMM_MY), B256, 0, stream>>>(
        m1, 2 * CC, W_m2, CC, out2, CC, b_m2, outb, CC, NN, CC, 2 * CC, ACT_NONE);
    // 16) out3 = relu(BN3(out2))  (in place)
    k_bnstats<<<CC, B256, 0, stream>>>(out2, mu, var);
    k_bnapply<<<ROWT, B256, 0, stream>>>(out2, mu, var, g3, be3, nullptr, 1, out2);
    // 17) pooled = mean over L  -> d_out[8:]
    k_pool<<<BB * CC / 256, B256, 0, stream>>>(out2, pooled);
    // 18) MLP head -> pred -> d_out[0:8]
    k_head<<<1, B256, 0, stream>>>(pooled, Wf1, bf1, Wf2, bf2, Wf3, bf3, pred);
}
